// SGCAFEModule_73624329388462
// MI455X (gfx1250) — matrix operand order repaired and checked by reading, not compiled, not run on hardware
//
#include <hip/hip_runtime.h>
#include <hip/hip_bf16.h>

// ---------------------------------------------------------------------------
// SGCAFE fused attention for MI455X (gfx1250, wave32, WMMA bf16 16x16x32).
//
// Roofline: ~88 GFLOP total vs ~0.5 GB HBM traffic (23.3 TB/s -> ~20us).
// bf16 WMMA (K=32/issue) keeps compute on the same order as the memory time;
// f32 WMMA (K=4) would be ~8x slower per issue.  All GEMMs run bf16 with f32
// accumulation; LN/softmax/sigmoid stay f32.
//
//  * V projection stores V^T ([B][INNER][N]) so the P@V B-fragment is a
//    pair of 16B loads per lane (was a 16-wide scalar gather).
//  * global_prefetch via __builtin_prefetch in the GEMM K-loops.
//  * mask row staged to LDS via async-to-LDS (builtin signature fixed: takes
//    int-vec16 pointers, global AS1 src / LDS AS3 dst).
// ---------------------------------------------------------------------------

#define B_SZ   32
#define N_TOK  576
#define C_DIM  1024
#define INNER  256
#define H_NUM  4
#define HD     64
#define M_TOK  (B_SZ * N_TOK)     // 18432 token rows
#define ALPHA  5.0f
#define EPS_LN 1e-5f

typedef __attribute__((ext_vector_type(16))) __bf16 v16bf;
typedef __attribute__((ext_vector_type(8)))  __bf16 v8bf;
typedef __attribute__((ext_vector_type(8)))  float  v8f;

__device__ __forceinline__ __bf16 f2bf(float f) {
  unsigned u = __float_as_uint(f);
  unsigned r = (u + 0x7FFFu + ((u >> 16) & 1u)) >> 16;   // RNE
  unsigned short s = (unsigned short)r;
  return __builtin_bit_cast(__bf16, s);
}

// 16-bit A-matrix 16x32 (MxK) layout (ISA 7.12.2): lane L<16 holds row L,
// K = {0..7, 16..23}; lane L+16 holds row L, K = {8..15, 24..31}.
__device__ __forceinline__ v16bf load_a_frag(const __bf16* base, int ld, int lane) {
  int row = lane & 15, half = lane >> 4;
  const __bf16* p = base + (size_t)row * ld + half * 8;
  v8bf lo = *reinterpret_cast<const v8bf*>(p);
  v8bf hi = *reinterpret_cast<const v8bf*>(p + 16);
  v16bf f;
#pragma unroll
  for (int i = 0; i < 8; i++) { f[i] = lo[i]; f[8 + i] = hi[i]; }
  return f;
}

// B-matrix 32x16 (KxN): lane holds column (lane&15); lanes 0-15 K={0..7,16..23},
// lanes 16-31 K={8..15,24..31} (matches A layout) -> two 16B loads.
__device__ __forceinline__ v16bf load_b_frag(const __bf16* base, int ld, int lane) {
  int col = lane & 15, half = lane >> 4;
  const __bf16* p = base + (size_t)col * ld + half * 8;
  v8bf lo = *reinterpret_cast<const v8bf*>(p);
  v8bf hi = *reinterpret_cast<const v8bf*>(p + 16);
  v16bf f;
#pragma unroll
  for (int i = 0; i < 8; i++) { f[i] = lo[i]; f[8 + i] = hi[i]; }
  return f;
}

__device__ __forceinline__ v8f wmma_bf16(v16bf a, v16bf b, v8f c) {
  return __builtin_amdgcn_wmma_f32_16x16x32_bf16(false, a, false, b, (short)0, c,
                                                 false, false);
}

__device__ __forceinline__ float blockReduceSum(float v, float* sh) {
  __syncthreads();
#pragma unroll
  for (int o = 16; o > 0; o >>= 1) v += __shfl_xor(v, o, 32);
  if ((threadIdx.x & 31) == 0) sh[threadIdx.x >> 5] = v;
  __syncthreads();
  float t = 0.f;
#pragma unroll
  for (int i = 0; i < 8; i++) t += sh[i];
  return t;
}

// --------------------------- Kernel 1: LayerNorms --------------------------
__global__ __launch_bounds__(256)
void ln_prep_kernel(const float* __restrict__ qf, const float* __restrict__ sf,
                    const float* __restrict__ lnqw, const float* __restrict__ lnqb,
                    const float* __restrict__ lnsw, const float* __restrict__ lnsb,
                    __bf16* __restrict__ qln, __bf16* __restrict__ sln,
                    __bf16* __restrict__ q16) {
  __shared__ float sh[8];
  int row = blockIdx.x, tid = threadIdx.x;
  const float* q = qf + (size_t)row * C_DIM;
  const float* s = sf + (size_t)row * C_DIM;
  float qv[4], sv[4], qs = 0.f, ss = 0.f;
#pragma unroll
  for (int i = 0; i < 4; i++) {
    qv[i] = q[tid + 256 * i]; sv[i] = s[tid + 256 * i];
    qs += qv[i]; ss += sv[i];
  }
  float mu_q = blockReduceSum(qs, sh) * (1.f / C_DIM);
  float mu_s = blockReduceSum(ss, sh) * (1.f / C_DIM);
  float q2 = 0.f, s2 = 0.f;
#pragma unroll
  for (int i = 0; i < 4; i++) {
    float a = qv[i] - mu_q; q2 += a * a;
    float b = sv[i] - mu_s; s2 += b * b;
  }
  float iq = rsqrtf(blockReduceSum(q2, sh) * (1.f / C_DIM) + EPS_LN);
  float is = rsqrtf(blockReduceSum(s2, sh) * (1.f / C_DIM) + EPS_LN);
#pragma unroll
  for (int i = 0; i < 4; i++) {
    int c = tid + 256 * i;
    size_t idx = (size_t)row * C_DIM + c;
    qln[idx] = f2bf((qv[i] - mu_q) * iq * lnqw[c] + lnqb[c]);
    sln[idx] = f2bf((sv[i] - mu_s) * is * lnsw[c] + lnsb[c]);
    q16[idx] = f2bf(qv[i]);
  }
}

// ------------------------ Kernel 2: weight conversion ----------------------
__global__ void cvt_bf16_kernel(const float* __restrict__ src,
                                __bf16* __restrict__ dst, int n) {
  int i = blockIdx.x * blockDim.x + threadIdx.x;
  int stride = gridDim.x * blockDim.x;
  for (; i < n; i += stride) dst[i] = f2bf(src[i]);
}

// ------------------ shared GEMM core: wave computes 16x32 ------------------
__device__ __forceinline__ void gemm_core(const __bf16* __restrict__ Xb,
                                          const __bf16* __restrict__ W0,
                                          const __bf16* __restrict__ W1,
                                          int K, int lane, v8f& acc0, v8f& acc1) {
  for (int k0 = 0; k0 < K; k0 += 32) {
    __builtin_prefetch(Xb + k0 + 128, 0, 1);           // global_prefetch_b8
    __builtin_prefetch(W0 + k0 + 128, 0, 1);
    v16bf a  = load_a_frag(Xb + k0, K, lane);          // reused for both tiles
    v16bf b0 = load_b_frag(W0 + k0, K, lane);
    v16bf b1 = load_b_frag(W1 + k0, K, lane);
    acc0 = wmma_bf16(a, b0, acc0);
    acc1 = wmma_bf16(a, b1, acc1);
  }
}

// -------------------- Kernel 3a: Y = X @ W^T + b  (bf16 out) ---------------
// block = 8 waves; wave (rw,cw) computes 16x32 at (row0+rw*16, col0+cw*32).
__global__ __launch_bounds__(256)
void gemm_xwt_bias_bf16_kernel(const __bf16* __restrict__ X,
                               const __bf16* __restrict__ W,
                               const float* __restrict__ bias,
                               __bf16* __restrict__ Y, int K, int NOUT) {
  int lane = threadIdx.x & 31, wid = threadIdx.x >> 5;
  int rw = wid & 3, cw = wid >> 2;
  int row0 = blockIdx.x * 64 + rw * 16;
  int col0 = blockIdx.y * 64 + cw * 32;
  v8f acc0 = {}, acc1 = {};
  gemm_core(X + (size_t)row0 * K, W + (size_t)col0 * K, W + (size_t)(col0 + 16) * K,
            K, lane, acc0, acc1);
  int col = lane & 15, half = lane >> 4;
  float bA = bias[col0 + col], bB = bias[col0 + 16 + col];
#pragma unroll
  for (int r = 0; r < 8; r++) {
    int row = row0 + r + 8 * half;                     // f32 C/D layout
    Y[(size_t)row * NOUT + col0 + col]      = f2bf(acc0[r] + bA);
    Y[(size_t)row * NOUT + col0 + 16 + col] = f2bf(acc1[r] + bB);
  }
}

// ------- Kernel 3b: Vt[b][inner][m] = (S @ Wv^T + bv)^T  (bf16 out) --------
// Same compute; epilogue stores transposed so the P@V B-fragment in the
// attention kernel is a pair of 16B loads.  Lane's 8 acc rows are 8
// consecutive tokens -> one 16B v8bf store per tile.
__global__ __launch_bounds__(256)
void gemm_xwt_bias_bf16_vt_kernel(const __bf16* __restrict__ X,
                                  const __bf16* __restrict__ W,
                                  const float* __restrict__ bias,
                                  __bf16* __restrict__ Vt, int K) {
  int lane = threadIdx.x & 31, wid = threadIdx.x >> 5;
  int rw = wid & 3, cw = wid >> 2;
  int row0 = blockIdx.x * 64 + rw * 16;                // token row (64 | 576)
  int col0 = blockIdx.y * 64 + cw * 32;                // inner col
  v8f acc0 = {}, acc1 = {};
  gemm_core(X + (size_t)row0 * K, W + (size_t)col0 * K, W + (size_t)(col0 + 16) * K,
            K, lane, acc0, acc1);
  int col = lane & 15, half = lane >> 4;
  float bA = bias[col0 + col], bB = bias[col0 + 16 + col];
  int b = row0 / N_TOK;
  int mbase = row0 - b * N_TOK + 8 * half;             // +r are consecutive m
  v8bf o0, o1;
#pragma unroll
  for (int r = 0; r < 8; r++) {
    o0[r] = f2bf(acc0[r] + bA);
    o1[r] = f2bf(acc1[r] + bB);
  }
  *reinterpret_cast<v8bf*>(Vt + ((size_t)(b * INNER + col0 + col)) * N_TOK + mbase) = o0;
  *reinterpret_cast<v8bf*>(Vt + ((size_t)(b * INNER + col0 + 16 + col)) * N_TOK + mbase) = o1;
}

// ----------------- Kernel 4: attention (one (b,h,16-row) tile) -------------
__global__ __launch_bounds__(128)
void attn_kernel(const __bf16* __restrict__ Qb, const __bf16* __restrict__ Kb,
                 const __bf16* __restrict__ Vt, const float* __restrict__ mask,
                 float* __restrict__ attn_out, __bf16* __restrict__ ctx) {
  constexpr int LLD = 577;               // f32 logits row stride
  constexpr int ALD = 592;               // bf16 probs row stride (32B aligned)
  __shared__ float  lg[16 * LLD];
  __shared__ __bf16 ap[16 * ALD];
  __shared__ float  msk[N_TOK];

  int lane = threadIdx.x & 31, wv = threadIdx.x >> 5;
  int ntile = blockIdx.x, bh = blockIdx.y;
  int b = bh >> 2, h = bh & 3;
  int btok = b * N_TOK;
  int tok0 = btok + ntile * 16;
  int col = lane & 15, half = lane >> 4;

  // Stage mask row into LDS (async-to-LDS path; signature per compiler:
  // (v4i __device__*, v4i LDS*, int offset, int cpol)).
  const float* mb = mask + (size_t)b * N_TOK;
#if defined(__gfx1250__) && __has_builtin(__builtin_amdgcn_global_load_async_to_lds_b128)
  {
    typedef int v4i __attribute__((vector_size(4 * sizeof(int))));
    typedef __attribute__((address_space(1))) v4i gv4i;
    typedef __attribute__((address_space(3))) v4i lv4i;
    int t = threadIdx.x;                                // 128 threads x 16B
    __builtin_amdgcn_global_load_async_to_lds_b128(
        (gv4i*)(uintptr_t)(mb + t * 4),
        (lv4i*)(uintptr_t)(unsigned)(size_t)(msk + t * 4), 0, 0);
    if (t < 16)                                         // remaining 64 floats
      __builtin_amdgcn_global_load_async_to_lds_b128(
          (gv4i*)(uintptr_t)(mb + 512 + t * 4),
          (lv4i*)(uintptr_t)(unsigned)(size_t)(msk + 512 + t * 4), 0, 0);
  }
#else
  for (int m = threadIdx.x; m < N_TOK; m += 128) msk[m] = mb[m];
#endif

  // Phase 1: logits = Q K^T / sqrt(HD); Q fragments hoisted across 36 tiles.
  v16bf aq0 = load_a_frag(Qb + (size_t)tok0 * INNER + h * HD +  0, INNER, lane);
  v16bf aq1 = load_a_frag(Qb + (size_t)tok0 * INNER + h * HD + 32, INNER, lane);
  for (int j = wv; j < N_TOK / 16; j += 4) {
    int m0 = j * 16;
    v8f acc = {};
    v16bf bk0 = load_b_frag(Kb + (size_t)(btok + m0) * INNER + h * HD +  0, INNER, lane);
    v16bf bk1 = load_b_frag(Kb + (size_t)(btok + m0) * INNER + h * HD + 32, INNER, lane);
    acc = wmma_bf16(aq0, bk0, acc);
    acc = wmma_bf16(aq1, bk1, acc);
#pragma unroll
    for (int r = 0; r < 8; r++)
      lg[(r + 8 * half) * LLD + m0 + col] = acc[r] * 0.125f;
  }
#if defined(__gfx1250__) && __has_builtin(__builtin_amdgcn_global_load_async_to_lds_b128)
#if __has_builtin(__builtin_amdgcn_s_wait_asynccnt)
  __builtin_amdgcn_s_wait_asynccnt(0);
#else
  asm volatile("s_wait_asynccnt 0" ::: "memory");
#endif
#endif
  __syncthreads();

  // Phase 2: +ALPHA*mask, softmax; f32 probs -> d_out, bf16 probs -> LDS.
  {
    int rrow = threadIdx.x >> 3;         // 16 rows x 8 threads
    int sub  = threadIdx.x & 7;
    float mx = -3.4e38f;
    for (int m = sub; m < N_TOK; m += 8) {
      float v = lg[rrow * LLD + m] + ALPHA * msk[m];
      lg[rrow * LLD + m] = v;
      mx = fmaxf(mx, v);
    }
#pragma unroll
    for (int o = 4; o > 0; o >>= 1) mx = fmaxf(mx, __shfl_xor(mx, o, 8));
    float sum = 0.f;
    for (int m = sub; m < N_TOK; m += 8) {
      float e = __expf(lg[rrow * LLD + m] - mx);
      lg[rrow * LLD + m] = e;
      sum += e;
    }
#pragma unroll
    for (int o = 4; o > 0; o >>= 1) sum += __shfl_xor(sum, o, 8);
    float inv = 1.f / sum;
    float* arow = attn_out + ((size_t)bh * N_TOK + ntile * 16 + rrow) * N_TOK;
    for (int m = sub; m < N_TOK; m += 8) {
      float p = lg[rrow * LLD + m] * inv;
      arow[m] = p;
      ap[rrow * ALD + m] = f2bf(p);
    }
  }
  __syncthreads();

  // Phase 3: ctx = P @ V; wave wv owns head-dim columns [wv*16, wv*16+16).
  // B operand reads V^T: lane col = d, contiguous in m -> two 16B loads.
  {
    v8f acc = {};
    int dcol0 = h * HD + wv * 16;
    const __bf16* VtBase = Vt + ((size_t)(b * INNER + dcol0)) * N_TOK;
    for (int m0 = 0; m0 < N_TOK; m0 += 32) {
      v16bf a  = load_a_frag(ap + m0, ALD, lane);
      v16bf bv = load_b_frag(VtBase + m0, N_TOK, lane);
      acc = wmma_bf16(a, bv, acc);
    }
#pragma unroll
    for (int r = 0; r < 8; r++)
      ctx[(size_t)(tok0 + r + 8 * half) * INNER + dcol0 + col] = f2bf(acc[r]);
  }
}

// --------- Kernel 5: out = ctx@Wo^T+bo, gate = sigmoid(q@Wg^T+bg), fuse ----
__global__ __launch_bounds__(256)
void fuse_kernel(const __bf16* __restrict__ ctxb, const __bf16* __restrict__ Wo16,
                 const float* __restrict__ bo, const __bf16* __restrict__ q16,
                 const __bf16* __restrict__ Wg16, const float* __restrict__ bg,
                 const float* __restrict__ qraw, float* __restrict__ enhanced) {
  int lane = threadIdx.x & 31, wid = threadIdx.x >> 5;
  int rw = wid >> 2, cw = wid & 3;
  int row0 = blockIdx.x * 32 + rw * 16;
  int col0 = blockIdx.y * 64 + cw * 16;
  v8f aco = {}, acg = {};
  const __bf16* Cb  = ctxb + (size_t)row0 * INNER;
  const __bf16* Wob = Wo16 + (size_t)col0 * INNER;
  for (int k0 = 0; k0 < INNER; k0 += 32) {
    v16bf a = load_a_frag(Cb + k0, INNER, lane);
    v16bf b = load_b_frag(Wob + k0, INNER, lane);
    aco = wmma_bf16(a, b, aco);
  }
  const __bf16* Qx  = q16 + (size_t)row0 * C_DIM;
  const __bf16* Wgb = Wg16 + (size_t)col0 * C_DIM;
  for (int k0 = 0; k0 < C_DIM; k0 += 32) {
    __builtin_prefetch(Qx + k0 + 128, 0, 1);
    __builtin_prefetch(Wgb + k0 + 128, 0, 1);
    v16bf a = load_a_frag(Qx + k0, C_DIM, lane);
    v16bf b = load_b_frag(Wgb + k0, C_DIM, lane);
    acg = wmma_bf16(a, b, acg);
  }
  int col = lane & 15, half = lane >> 4;
  int c = col0 + col;
  float bO = bo[c], bG = bg[c];
#pragma unroll
  for (int r = 0; r < 8; r++) {
    int row = row0 + r + 8 * half;
    float o = aco[r] + bO;
    float g = 1.f / (1.f + __expf(-(acg[r] + bG)));
    float qv = qraw[(size_t)row * C_DIM + c];
    enhanced[(size_t)row * C_DIM + c] = qv + g * o;
  }
}

// ---------------------------------------------------------------------------
extern "C" void kernel_launch(void* const* d_in, const int* in_sizes, int n_in,
                              void* d_out, int out_size, void* d_ws, size_t ws_size,
                              hipStream_t stream) {
  const float* qf   = (const float*)d_in[0];
  const float* sf   = (const float*)d_in[1];
  const float* mask = (const float*)d_in[2];
  const float* lnqw = (const float*)d_in[3];
  const float* lnqb = (const float*)d_in[4];
  const float* lnsw = (const float*)d_in[5];
  const float* lnsb = (const float*)d_in[6];
  const float* Wq = (const float*)d_in[7];  const float* bq = (const float*)d_in[8];
  const float* Wk = (const float*)d_in[9];  const float* bk = (const float*)d_in[10];
  const float* Wv = (const float*)d_in[11]; const float* bv = (const float*)d_in[12];
  const float* Wo = (const float*)d_in[13]; const float* bo = (const float*)d_in[14];
  const float* Wg = (const float*)d_in[15]; const float* bg = (const float*)d_in[16];

  float* enhanced = (float*)d_out;
  float* attn_out = enhanced + (size_t)M_TOK * C_DIM;

  char* ws = (char*)d_ws;
  size_t off = 0;
  auto carve = [&](size_t bytes) -> char* {
    char* p = ws + off;
    off = (off + bytes + 255) & ~(size_t)255;
    return p;
  };
  __bf16* qln  = (__bf16*)carve((size_t)M_TOK * C_DIM * 2);
  __bf16* sln  = (__bf16*)carve((size_t)M_TOK * C_DIM * 2);
  __bf16* q16  = (__bf16*)carve((size_t)M_TOK * C_DIM * 2);
  __bf16* Wq16 = (__bf16*)carve((size_t)INNER * C_DIM * 2);
  __bf16* Wk16 = (__bf16*)carve((size_t)INNER * C_DIM * 2);
  __bf16* Wv16 = (__bf16*)carve((size_t)INNER * C_DIM * 2);
  __bf16* Wo16 = (__bf16*)carve((size_t)C_DIM * INNER * 2);
  __bf16* Wg16 = (__bf16*)carve((size_t)C_DIM * C_DIM * 2);
  __bf16* Qb   = (__bf16*)carve((size_t)M_TOK * INNER * 2);
  __bf16* Kb   = (__bf16*)carve((size_t)M_TOK * INNER * 2);
  __bf16* Vt   = (__bf16*)carve((size_t)B_SZ * INNER * N_TOK * 2);
  __bf16* ctxb = (__bf16*)carve((size_t)M_TOK * INNER * 2);

  ln_prep_kernel<<<M_TOK, 256, 0, stream>>>(qf, sf, lnqw, lnqb, lnsw, lnsb,
                                            qln, sln, q16);
  cvt_bf16_kernel<<<512, 256, 0, stream>>>(Wq, Wq16, INNER * C_DIM);
  cvt_bf16_kernel<<<512, 256, 0, stream>>>(Wk, Wk16, INNER * C_DIM);
  cvt_bf16_kernel<<<512, 256, 0, stream>>>(Wv, Wv16, INNER * C_DIM);
  cvt_bf16_kernel<<<512, 256, 0, stream>>>(Wo, Wo16, C_DIM * INNER);
  cvt_bf16_kernel<<<2048, 256, 0, stream>>>(Wg, Wg16, C_DIM * C_DIM);

  dim3 gqkv(M_TOK / 64, INNER / 64);
  gemm_xwt_bias_bf16_kernel<<<gqkv, 256, 0, stream>>>(qln, Wq16, bq, Qb, C_DIM, INNER);
  gemm_xwt_bias_bf16_kernel<<<gqkv, 256, 0, stream>>>(sln, Wk16, bk, Kb, C_DIM, INNER);
  gemm_xwt_bias_bf16_vt_kernel<<<gqkv, 256, 0, stream>>>(sln, Wv16, bv, Vt, C_DIM);

  attn_kernel<<<dim3(N_TOK / 16, B_SZ * H_NUM), 128, 0, stream>>>(Qb, Kb, Vt, mask,
                                                                  attn_out, ctxb);

  fuse_kernel<<<dim3(M_TOK / 32, C_DIM / 64), 256, 0, stream>>>(ctxb, Wo16, bo, q16,
                                                                Wg16, bg, qf, enhanced);
}
